// LSTM_68874095559126
// MI455X (gfx1250) — matrix operand order repaired and checked by reading, not compiled, not run on hardware
//
#include <hip/hip_runtime.h>
#include <hip/hip_bf16.h>

#define BB   256
#define TT   1024
#define HH   100
#define CC   9
#define GG   400      // 4*H
#define NCls 6
#define K1P  128      // padded K for layer1 ([x|h1], real 109)
#define CH1  4        // K chunks of 32
#define K2P  224      // padded K for layer2 ([h1|h2], real 200)
#define CH2  7
#define NT   25       // N tiles (400/16)
#define NW   8        // waves per block
#define NTHR 256

typedef __attribute__((ext_vector_type(16))) __bf16 v16bf;
typedef __attribute__((ext_vector_type(8)))  float  v8f;

// ---- LDS layout (all offsets in elements) ----
#define W1SZ (NT * CH1 * 32 * 16)   // 51200 ushort
#define W2SZ (NT * CH2 * 32 * 16)   // 89600 ushort
#define A1SZ (CH1 * 32 * 16)        // 2048 ushort
#define A2SZ (CH2 * 32 * 16)        // 3584 ushort
#define SMEM_BYTES ((W1SZ + W2SZ + A1SZ + A2SZ) * 2 + (16 * GG + GG + GG + NCls * HH + NCls) * 4) // 324088

__device__ __forceinline__ unsigned short f2bf(float f) {
  union { float f; unsigned u; } c; c.f = f;
  unsigned u = c.u;
  u += 0x7fffu + ((u >> 16) & 1u);       // round-to-nearest-even
  return (unsigned short)(u >> 16);
}
__device__ __forceinline__ float bf2f(unsigned short h) {
  union { unsigned u; float f; } c; c.u = ((unsigned)h) << 16;
  return c.f;
}
__device__ __forceinline__ float sigm(float x) { return 1.0f / (1.0f + __expf(-x)); }

// ushort index of A-matrix element (k, m) within a fragment array laid out as
// [chunk][lane][16 elems] matching V_WMMA 16x16x32 bf16 A layout.
__device__ __forceinline__ int aslot(int k, int m) {
  int chunk = k >> 5, kk = k & 31;
  int lane  = (((kk >> 3) & 1) << 4) | m;
  int elem  = ((((kk >> 4) << 2) | ((kk >> 1) & 3)) << 1) | (kk & 1);
  return (chunk * 32 + lane) * 16 + elem;
}
// ushort index of B-matrix element (k, nn) within one tile's fragment region.
__device__ __forceinline__ int bslot(int chunk, int kk, int nn) {
  int lane = (((kk >> 3) & 1) << 4) | nn;
  return (chunk * 32 + lane) * 16 + (((kk >> 4) << 3) | (kk & 7));
}

template <int CH>
__device__ __forceinline__ void gemm_phase(const unsigned short* __restrict__ Wf,
                                           const unsigned short* __restrict__ Af,
                                           const float* __restrict__ bias,
                                           float* __restrict__ gates,
                                           int ntile, int wv, int lane) {
  v8f z = {0.f, 0.f, 0.f, 0.f, 0.f, 0.f, 0.f, 0.f};
  v8f acc[4];
#pragma unroll
  for (int s = 0; s < 4; ++s) acc[s] = z;
#pragma unroll
  for (int ch = 0; ch < CH; ++ch) {
    v16bf a = *(const v16bf*)(Af + (ch * 32 + lane) * 16);
#pragma unroll
    for (int s = 0; s < 4; ++s) {
      if (s < ntile) {
        int n = wv + s * NW;
        v16bf bm = *(const v16bf*)(Wf + ((n * CH + ch) * 32 + lane) * 16);
        acc[s] = __builtin_amdgcn_wmma_f32_16x16x32_bf16(
            false, a, false, bm, (short)0, acc[s], false, false);
      }
    }
  }
  int mb = (lane >> 4) << 3;                     // M base: 0 or 8
  for (int s = 0; s < ntile; ++s) {
    int n   = wv + s * NW;
    int col = (n << 4) + (lane & 15);
    float bv = bias[col];
#pragma unroll
    for (int r = 0; r < 8; ++r)
      gates[(mb + r) * GG + col] = acc[s][r] + bv;
  }
}

__global__ void __launch_bounds__(NTHR, 1)
lstm_fused(const float* __restrict__ x,
           const float* __restrict__ h01, const float* __restrict__ c01,
           const float* __restrict__ h02, const float* __restrict__ c02,
           const float* __restrict__ Wih1, const float* __restrict__ Whh1,
           const float* __restrict__ b1,
           const float* __restrict__ Wih2, const float* __restrict__ Whh2,
           const float* __restrict__ b2,
           const float* __restrict__ Wfc, const float* __restrict__ bfc_g,
           float* __restrict__ out) {
  extern __shared__ __align__(16) char smem[];
  unsigned short* W1 = (unsigned short*)smem;
  unsigned short* W2 = W1 + W1SZ;
  unsigned short* A1 = W2 + W2SZ;
  unsigned short* A2 = A1 + A1SZ;
  float* gates = (float*)(A2 + A2SZ);
  float* b1s   = gates + 16 * GG;
  float* b2s   = b1s + GG;
  float* wfcs  = b2s + GG;
  float* bfcs  = wfcs + NCls * HH;

  const int tid  = threadIdx.x;
  const int lane = tid & 31;
  const int wv   = tid >> 5;
  const int b0   = blockIdx.x << 4;               // 16 batch rows per WG
  const int ntile = 3 + (wv == 0 ? 1 : 0);        // tiles wv, wv+8, wv+16 (+24 for wv 0)

  // ---- one-time: swizzle weights into WMMA B-fragment layout (zero-padded K) ----
  for (int idx = tid; idx < GG * K1P; idx += NTHR) {
    int col = idx / K1P, k = idx - col * K1P;
    float v = 0.f;
    if (k < CC)            v = Wih1[col * CC + k];
    else if (k < CC + HH)  v = Whh1[col * HH + (k - CC)];
    W1[(col >> 4) * (CH1 * 32 * 16) + bslot(k >> 5, k & 31, col & 15)] = f2bf(v);
  }
  for (int idx = tid; idx < GG * K2P; idx += NTHR) {
    int col = idx / K2P, k = idx - col * K2P;
    float v = 0.f;
    if (k < HH)            v = Wih2[col * HH + k];
    else if (k < 2 * HH)   v = Whh2[col * HH + (k - HH)];
    W2[(col >> 4) * (CH2 * 32 * 16) + bslot(k >> 5, k & 31, col & 15)] = f2bf(v);
  }
  for (int i = tid; i < GG; i += NTHR) { b1s[i] = b1[i]; b2s[i] = b2[i]; }
  for (int i = tid; i < NCls * HH; i += NTHR) wfcs[i] = Wfc[i];
  if (tid < NCls) bfcs[tid] = bfc_g[tid];

  // zero A-matrix K padding
  for (int i = tid; i < 16 * (K1P - (CC + HH)); i += NTHR)
    A1[aslot((CC + HH) + (i >> 4), i & 15)] = 0;
  for (int i = tid; i < 16 * (K2P - 2 * HH); i += NTHR)
    A2[aslot(2 * HH + (i >> 4), i & 15)] = 0;

  // cell state lives in registers; init h0 into A fragments
  float c1r[7], c2r[7];
#pragma unroll
  for (int r = 0; r < 7; ++r) {
    int q = tid + r * NTHR;
    if (q < 16 * HH) {
      int m = q & 15, j = q >> 4;
      c1r[r] = c01[(b0 + m) * HH + j];
      c2r[r] = c02[(b0 + m) * HH + j];
      A1[aslot(CC + j, m)] = f2bf(h01[(b0 + m) * HH + j]);
      A2[aslot(HH + j, m)] = f2bf(h02[(b0 + m) * HH + j]);
    } else { c1r[r] = 0.f; c2r[r] = 0.f; }
  }
  __syncthreads();

  for (int t = 0; t < TT; ++t) {
    // phase A: stage x_t (16 rows x 9 channels) into A1 k=0..8
    if (tid < 16 * CC) {
      int m = tid / CC, ch = tid - m * CC;
      const float* px = &x[((b0 + m) * CC + ch) * TT + t];
      A1[aslot(ch, m)] = f2bf(*px);
      if (t + 1 < TT) __builtin_prefetch(px + 1, 0, 3);
    }
    __syncthreads();

    // phase B: layer-1 gates = [x_t | h1] x W1^T + b1   (WMMA)
    gemm_phase<CH1>(W1, A1, b1s, gates, ntile, wv, lane);
    __syncthreads();

    // phase C: layer-1 cell update; h1 feeds A1 (next step) and A2 (this step)
#pragma unroll
    for (int r = 0; r < 7; ++r) {
      int q = tid + r * NTHR;
      if (q < 16 * HH) {
        int m = q & 15, j = q >> 4;
        float gi = gates[m * GG + j];
        float gf = gates[m * GG + HH + j];
        float gc = gates[m * GG + 2 * HH + j];
        float go = gates[m * GG + 3 * HH + j];
        float c  = sigm(gf) * c1r[r] + sigm(gi) * tanhf(gc);
        c1r[r]   = c;
        unsigned short hb = f2bf(sigm(go) * tanhf(c));
        A1[aslot(CC + j, m)] = hb;
        A2[aslot(j, m)]      = hb;
      }
    }
    __syncthreads();

    // phase D: layer-2 gates = [h1_t | h2] x W2^T + b2   (WMMA)
    gemm_phase<CH2>(W2, A2, b2s, gates, ntile, wv, lane);
    __syncthreads();

    // phase E: layer-2 cell update; h2 back into A2 k=100..199
#pragma unroll
    for (int r = 0; r < 7; ++r) {
      int q = tid + r * NTHR;
      if (q < 16 * HH) {
        int m = q & 15, j = q >> 4;
        float gi = gates[m * GG + j];
        float gf = gates[m * GG + HH + j];
        float gc = gates[m * GG + 2 * HH + j];
        float go = gates[m * GG + 3 * HH + j];
        float c  = sigm(gf) * c2r[r] + sigm(gi) * tanhf(gc);
        c2r[r]   = c;
        A2[aslot(HH + j, m)] = f2bf(sigm(go) * tanhf(c));
      }
    }
    __syncthreads();
  }

  // final FC: out[m][n] = h2_last . Wfc[n] + bfc   (16x6 outputs)
  if (tid < 16 * NCls) {
    int m = tid / NCls, n = tid - m * NCls;
    float s = bfcs[n];
    for (int j = 0; j < HH; ++j)
      s += bf2f(A2[aslot(HH + j, m)]) * wfcs[n * HH + j];
    out[(b0 + m) * NCls + n] = s;
  }
}

extern "C" void kernel_launch(void* const* d_in, const int* in_sizes, int n_in,
                              void* d_out, int out_size, void* d_ws, size_t ws_size,
                              hipStream_t stream) {
  (void)in_sizes; (void)n_in; (void)out_size; (void)d_ws; (void)ws_size;
  const float* x    = (const float*)d_in[0];
  const float* h01  = (const float*)d_in[1];
  const float* c01  = (const float*)d_in[2];
  const float* h02  = (const float*)d_in[3];
  const float* c02  = (const float*)d_in[4];
  const float* Wih1 = (const float*)d_in[5];
  const float* Whh1 = (const float*)d_in[6];
  const float* b1   = (const float*)d_in[7];
  const float* Wih2 = (const float*)d_in[8];
  const float* Whh2 = (const float*)d_in[9];
  const float* b2   = (const float*)d_in[10];
  const float* Wfc  = (const float*)d_in[11];
  const float* bfc  = (const float*)d_in[12];
  float* out = (float*)d_out;

  static int attr_done = 0;   // idempotent attribute set (not a stream op; capture-safe)
  if (!attr_done) {
    hipFuncSetAttribute((const void*)lstm_fused,
                        hipFuncAttributeMaxDynamicSharedMemorySize, SMEM_BYTES);
    attr_done = 1;
  }
  lstm_fused<<<dim3(BB / 16), dim3(NTHR), SMEM_BYTES, stream>>>(
      x, h01, c01, h02, c02, Wih1, Whh1, b1, Wih2, Whh2, b2, Wfc, bfc, out);
}
